// MSDeformAttn_1159641170354
// MI455X (gfx1250) — matrix operand order repaired and checked by reading, not compiled, not run on hardware
//
#include <hip/hip_runtime.h>

// ---------------- Problem constants ----------------
#define D_MODEL   256
#define N_HEADS   8
#define N_LEVELS  4
#define N_POINTS  4
#define HEAD_DIM  32
#define BATCH     4
#define LQ        5440            // 64*64 + 32*32 + 16*16 + 8*8
#define ROWS      (BATCH * LQ)    // 21760
#define NQ_ELEMS  (ROWS * D_MODEL) // 5,570,560

// level geometry (compile-time)
__device__ __constant__ const int c_H[4]  = {64, 32, 16, 8};
__device__ __constant__ const int c_W[4]  = {64, 32, 16, 8};
__device__ __constant__ const int c_LS[4] = {0, 4096, 5120, 5376};

// ---------------- Types ----------------
typedef __attribute__((ext_vector_type(16))) __bf16 v16bf;
typedef __attribute__((ext_vector_type(2)))  __bf16 v2bf;
typedef __attribute__((ext_vector_type(8)))  float  v8f;
typedef unsigned short u16;

// ---------------- bf16 helpers ----------------
__device__ __forceinline__ u16 f2bf_u(float f) {
  unsigned u = __builtin_bit_cast(unsigned, f);
  unsigned r = u + 0x7FFFu + ((u >> 16) & 1u);      // round-to-nearest-even
  return (u16)(r >> 16);
}
__device__ __forceinline__ float bf2f(u16 h) {
  return __builtin_bit_cast(float, (unsigned)h << 16);
}

// ----------------------------------------------------------------------------
// Wave-level WMMA GEMM: one wave computes a 16x64 tile of C = A * B^T_layout.
// A  : row-major [rows][256] bf16 (stored as u16)
// Bt : row-major [cols][256] bf16 (i.e. original B transposed so K is innermost)
// acc: 4 accumulators, each a 16x16 f32 C/D fragment.
//
// CDNA5 VGPR layouts (ISA 7.12.2):
//   A 16x32 bf16 : M = lane%16; VGPR v<4 -> K = half*8 + 2v (+p); v>=4 -> K = 16 + half*8 + 2(v-4)
//   B 32x16 bf16 : N = lane%16; VGPR v<4 -> K = half*8 + 2v (+p); v>=4 -> K = 16 + half*8 + 2(v-4)
//   C 16x16 f32  : M = r + 8*half; N = lane%16
// ----------------------------------------------------------------------------
__device__ __forceinline__ void wmma_wave_16x64(const u16* __restrict__ A,
                                                const u16* __restrict__ Bt,
                                                int rowbase, int colbase,
                                                v8f acc[4]) {
  const int lane = threadIdx.x & 31;
  const int half = lane >> 4;
  const int lr   = lane & 15;
  const u16* arow = A + (size_t)(rowbase + lr) * D_MODEL;

  for (int k0 = 0; k0 < D_MODEL; k0 += 32) {
    if (k0 + 32 < D_MODEL) __builtin_prefetch(arow + k0 + 32, 0, 1); // global_prefetch_b8
    v16bf a;
#pragma unroll
    for (int v = 0; v < 8; ++v) {
      const int kb = (v < 4) ? (half * 8 + v * 2) : (16 + half * 8 + (v - 4) * 2);
      v2bf pr = *(const v2bf*)(arow + k0 + kb);
      a[2 * v]     = pr.x;
      a[2 * v + 1] = pr.y;
    }
#pragma unroll
    for (int t = 0; t < 4; ++t) {
      const u16* brow = Bt + (size_t)(colbase + t * 16 + lr) * D_MODEL;
      v16bf bb;
#pragma unroll
      for (int v = 0; v < 8; ++v) {
        const int kb = (v < 4) ? (half * 8 + v * 2) : (16 + half * 8 + (v - 4) * 2);
        v2bf pr = *(const v2bf*)(brow + k0 + kb);
        bb[2 * v]     = pr.x;
        bb[2 * v + 1] = pr.y;
      }
      acc[t] = __builtin_amdgcn_wmma_f32_16x16x32_bf16(
          false, a, false, bb, (short)0, acc[t], false, false);
    }
  }
}

// ----------------------------------------------------------------------------
// Kernel 1: precision prep. bf16 copies of activations + transposed bf16 weights.
// ----------------------------------------------------------------------------
__global__ void __launch_bounds__(256) prep_kernel(
    const float* __restrict__ query, const float* __restrict__ flat,
    const float* __restrict__ W_val, const float* __restrict__ W_off,
    const float* __restrict__ W_attn, const float* __restrict__ W_out,
    u16* __restrict__ qbf, u16* __restrict__ fbf,
    u16* __restrict__ Wvt, u16* __restrict__ Wot,
    u16* __restrict__ Wat, u16* __restrict__ Wut) {
  const int i = blockIdx.x * 256 + threadIdx.x;
  if (i < NQ_ELEMS) {
    qbf[i] = f2bf_u(query[i]);
    fbf[i] = f2bf_u(flat[i]);
  }
  if (i < D_MODEL * D_MODEL) {           // 65536: Wt[n][k] = W[k][n]
    const int n = i >> 8, k = i & 255;
    Wvt[i] = f2bf_u(W_val[k * D_MODEL + n]);
    Wot[i] = f2bf_u(W_off[k * D_MODEL + n]);
    Wut[i] = f2bf_u(W_out[k * D_MODEL + n]);
  }
  if (i < D_MODEL * 128) {               // 32768: W_attn is 256x128
    const int n = i >> 8, k = i & 255;
    Wat[i] = f2bf_u(W_attn[k * 128 + n]);
  }
}

// ----------------------------------------------------------------------------
// Kernel 2: value = input_flatten @ W_val + b_val, stored bf16 as [b][m][hw][32]
// grid (340, 2), block 256 (8 waves). Block tile 64 rows x 128 cols.
// ----------------------------------------------------------------------------
__global__ void __launch_bounds__(256) value_proj_kernel(
    const u16* __restrict__ fbf, const u16* __restrict__ Wvt,
    const float* __restrict__ b_val, u16* __restrict__ valbf) {
  const int wave = threadIdx.x >> 5, lane = threadIdx.x & 31;
  const int half = lane >> 4, lr = lane & 15;
  const int rowbase = blockIdx.x * 64 + (wave & 3) * 16;
  const int cb      = blockIdx.y * 128 + (wave >> 2) * 64;
  v8f acc[4] = {};
  wmma_wave_16x64(fbf, Wvt, rowbase, cb, acc);
#pragma unroll
  for (int t = 0; t < 4; ++t) {
    const int c = cb + t * 16 + lr;
    const int m = c >> 5, d = c & 31;
#pragma unroll
    for (int r = 0; r < 8; ++r) {
      const int row = rowbase + r + 8 * half;      // = b*LQ + n
      const int b = row / LQ, n = row - b * LQ;
      const float v = acc[t][r] + b_val[c];
      valbf[(((size_t)(b * N_HEADS + m)) * LQ + n) * HEAD_DIM + d] = f2bf_u(v);
    }
  }
}

// ----------------------------------------------------------------------------
// Kernel 3: OFF = query @ W_off + b_off (f32), ATTN = softmax(query @ W_attn + b_attn)
// grid (340, 3): group 0/1 -> offset cols [0,128)/[128,256); group 2 -> all 128 attn
// cols in one block so softmax (16-wide groups) is block-local via LDS.
// ----------------------------------------------------------------------------
__global__ void __launch_bounds__(256) offattn_kernel(
    const u16* __restrict__ qbf,
    const u16* __restrict__ Wot, const float* __restrict__ b_off,
    const u16* __restrict__ Wat, const float* __restrict__ b_attn,
    float* __restrict__ OFF, float* __restrict__ ATTN) {
  __shared__ float logits[64][128];
  const int wave = threadIdx.x >> 5, lane = threadIdx.x & 31;
  const int half = lane >> 4, lr = lane & 15;
  const int rowbase = blockIdx.x * 64 + (wave & 3) * 16;
  const int group = blockIdx.y;

  if (group < 2) {
    const int cb = group * 128 + (wave >> 2) * 64;
    v8f acc[4] = {};
    wmma_wave_16x64(qbf, Wot, rowbase, cb, acc);
#pragma unroll
    for (int t = 0; t < 4; ++t) {
      const int c = cb + t * 16 + lr;
#pragma unroll
      for (int r = 0; r < 8; ++r) {
        const int row = rowbase + r + 8 * half;
        OFF[(size_t)row * D_MODEL + c] = acc[t][r] + b_off[c];
      }
    }
  } else {
    const int cb = (wave >> 2) * 64;               // 0 or 64, covers 128 attn cols
    v8f acc[4] = {};
    wmma_wave_16x64(qbf, Wat, rowbase, cb, acc);
#pragma unroll
    for (int t = 0; t < 4; ++t) {
      const int c = cb + t * 16 + lr;
#pragma unroll
      for (int r = 0; r < 8; ++r) {
        const int rlocal = (wave & 3) * 16 + r + 8 * half;
        logits[rlocal][c] = acc[t][r] + b_attn[c];
      }
    }
    __syncthreads();
    // softmax over 16 logits per (row, head): 64*8 = 512 groups, 256 threads
    for (int it = threadIdx.x; it < 512; it += 256) {
      const int rl = it >> 3, mh = it & 7;
      const float* lp = &logits[rl][mh * 16];
      float mx = lp[0];
#pragma unroll
      for (int j = 1; j < 16; ++j) mx = fmaxf(mx, lp[j]);
      float e[16], sum = 0.f;
#pragma unroll
      for (int j = 0; j < 16; ++j) { e[j] = __expf(lp[j] - mx); sum += e[j]; }
      const float inv = 1.0f / sum;
      const size_t base = ((size_t)(blockIdx.x * 64 + rl)) * 128 + mh * 16;
#pragma unroll
      for (int j = 0; j < 16; ++j) ATTN[base + j] = e[j] * inv;
    }
  }
}

// ----------------------------------------------------------------------------
// Kernel 4: deformable bilinear sampling. One wave32 per (b, q, m); lane = channel d.
// Each corner fetch is a coalesced 64B bf16 load (value is [b][m][hw][32]).
// Exactly matches torch grid_sample(bilinear, zeros, align_corners=False).
// ----------------------------------------------------------------------------
__device__ __forceinline__ float corner_val(const u16* __restrict__ v,
                                            int xi, int yi, int W, int H, int lane) {
  if (xi < 0 || xi >= W || yi < 0 || yi >= H) return 0.0f;
  return bf2f(v[(size_t)(yi * W + xi) * HEAD_DIM + lane]);
}

__global__ void __launch_bounds__(256) sample_kernel(
    const float* __restrict__ refpts, const float* __restrict__ OFF,
    const float* __restrict__ ATTN, const u16* __restrict__ valbf,
    u16* __restrict__ headbf) {
  const int wid  = (blockIdx.x * 256 + threadIdx.x) >> 5;  // (b*LQ+q)*8 + m
  const int lane = threadIdx.x & 31;
  const int m  = wid & 7;
  const int bq = wid >> 3;                 // b*LQ + q
  const int b  = bq / LQ;

  const float* ref     = refpts + (size_t)bq * (N_LEVELS * 2);
  const float* offrow  = OFF + (size_t)bq * D_MODEL + m * 32;   // (l*4+p)*2 + xy
  const float* attnrow = ATTN + (size_t)bq * 128 + m * 16;      // l*4 + p
  const u16*   vbase   = valbf + ((size_t)(b * N_HEADS + m)) * LQ * HEAD_DIM;

  float acc = 0.0f;
#pragma unroll
  for (int l = 0; l < N_LEVELS; ++l) {
    const int H = c_H[l], W = c_W[l];
    const float rx = ref[l * 2 + 0], ry = ref[l * 2 + 1];
    const u16* vlvl = vbase + (size_t)c_LS[l] * HEAD_DIM;
    const float invW = 1.0f / (float)W, invH = 1.0f / (float)H;
#pragma unroll
    for (int p = 0; p < N_POINTS; ++p) {
      const float ox = offrow[(l * 4 + p) * 2 + 0];
      const float oy = offrow[(l * 4 + p) * 2 + 1];
      const float aw = attnrow[l * 4 + p];
      // loc = ref + off / (W,H); grid = 2*loc - 1; x = (grid+1)*W/2 - 0.5 = loc*W - 0.5
      const float x = (rx + ox * invW) * (float)W - 0.5f;
      const float y = (ry + oy * invH) * (float)H - 0.5f;
      const float x0f = floorf(x), y0f = floorf(y);
      const int   x0 = (int)x0f,  y0 = (int)y0f;
      const float wx1 = x - x0f, wx0 = 1.0f - wx1;
      const float wy1 = y - y0f, wy0 = 1.0f - wy1;
      float s;
      s  = corner_val(vlvl, x0,     y0,     W, H, lane) * (wy0 * wx0);
      s += corner_val(vlvl, x0 + 1, y0,     W, H, lane) * (wy0 * wx1);
      s += corner_val(vlvl, x0,     y0 + 1, W, H, lane) * (wy1 * wx0);
      s += corner_val(vlvl, x0 + 1, y0 + 1, W, H, lane) * (wy1 * wx1);
      acc += aw * s;
    }
  }
  headbf[(size_t)bq * D_MODEL + m * HEAD_DIM + lane] = f2bf_u(acc);
}

// ----------------------------------------------------------------------------
// Kernel 5: out = head @ W_out + b_out (f32 output). grid (340, 2).
// ----------------------------------------------------------------------------
__global__ void __launch_bounds__(256) out_proj_kernel(
    const u16* __restrict__ headbf, const u16* __restrict__ Wut,
    const float* __restrict__ b_out, float* __restrict__ out) {
  const int wave = threadIdx.x >> 5, lane = threadIdx.x & 31;
  const int half = lane >> 4, lr = lane & 15;
  const int rowbase = blockIdx.x * 64 + (wave & 3) * 16;
  const int cb      = blockIdx.y * 128 + (wave >> 2) * 64;
  v8f acc[4] = {};
  wmma_wave_16x64(headbf, Wut, rowbase, cb, acc);
#pragma unroll
  for (int t = 0; t < 4; ++t) {
    const int c = cb + t * 16 + lr;
#pragma unroll
    for (int r = 0; r < 8; ++r) {
      const int row = rowbase + r + 8 * half;
      out[(size_t)row * D_MODEL + c] = acc[t][r] + b_out[c];
    }
  }
}

// ----------------------------------------------------------------------------
// Host launcher
// ----------------------------------------------------------------------------
extern "C" void kernel_launch(void* const* d_in, const int* in_sizes, int n_in,
                              void* d_out, int out_size, void* d_ws, size_t ws_size,
                              hipStream_t stream) {
  (void)in_sizes; (void)n_in; (void)out_size; (void)ws_size;
  const float* query  = (const float*)d_in[0];
  const float* refpts = (const float*)d_in[1];
  const float* flat   = (const float*)d_in[2];
  const float* W_off  = (const float*)d_in[3];
  const float* b_off  = (const float*)d_in[4];
  const float* W_attn = (const float*)d_in[5];
  const float* b_attn = (const float*)d_in[6];
  const float* W_val  = (const float*)d_in[7];
  const float* b_val  = (const float*)d_in[8];
  const float* W_out  = (const float*)d_in[9];
  const float* b_out  = (const float*)d_in[10];
  float* out = (float*)d_out;

  // workspace layout (all offsets 256B-aligned)
  char* ws = (char*)d_ws;
  size_t o = 0;
  float* OFF    = (float*)(ws + o); o += (size_t)ROWS * 256 * 4;  // 22,282,240
  float* ATTN   = (float*)(ws + o); o += (size_t)ROWS * 128 * 4;  // 11,141,120
  u16*   qbf    = (u16*)  (ws + o); o += (size_t)NQ_ELEMS * 2;
  u16*   fbf    = (u16*)  (ws + o); o += (size_t)NQ_ELEMS * 2;
  u16*   valbf  = (u16*)  (ws + o); o += (size_t)NQ_ELEMS * 2;
  u16*   headbf = (u16*)  (ws + o); o += (size_t)NQ_ELEMS * 2;
  u16*   Wvt    = (u16*)  (ws + o); o += 65536 * 2;
  u16*   Wot    = (u16*)  (ws + o); o += 65536 * 2;
  u16*   Wat    = (u16*)  (ws + o); o += 32768 * 2;
  u16*   Wut    = (u16*)  (ws + o); o += 65536 * 2;

  const int nblk = NQ_ELEMS / 256;                 // 21760
  prep_kernel<<<nblk, 256, 0, stream>>>(query, flat, W_val, W_off, W_attn, W_out,
                                        qbf, fbf, Wvt, Wot, Wat, Wut);
  value_proj_kernel<<<dim3(ROWS / 64, 2), 256, 0, stream>>>(fbf, Wvt, b_val, valbf);
  offattn_kernel<<<dim3(ROWS / 64, 3), 256, 0, stream>>>(qbf, Wot, b_off, Wat, b_attn,
                                                         OFF, ATTN);
  sample_kernel<<<nblk, 256, 0, stream>>>(refpts, OFF, ATTN, valbf, headbf);
  out_proj_kernel<<<dim3(ROWS / 64, 2), 256, 0, stream>>>(headbf, Wut, b_out, out);
}
